// GAT_75926431859395
// MI455X (gfx1250) — matrix operand order repaired and checked by reading, not compiled, not run on hardware
//
#include <hip/hip_runtime.h>
#include <hip/hip_bf16.h>

// ---------------------------------------------------------------------------
// GATv2 (2 layers) + mean-pool + MLP head for gfx1250 (MI455X, wave32, WMMA)
// ---------------------------------------------------------------------------

#define NEG_SLOPE 0.2f

typedef __attribute__((ext_vector_type(16))) __bf16 v16bf;
typedef __attribute__((ext_vector_type(8)))  float  v8f;

// ------------------------------ helpers ------------------------------------

__device__ __forceinline__ void get_edge(const long long* __restrict__ ei,
                                         int E, int e, int& s, int& d) {
    if (e < E) { s = (int)ei[e]; d = (int)ei[(size_t)E + e]; }
    else       { s = d = e - E; }               // self loop
}

__device__ __forceinline__ unsigned enc_ordered(float f) {
    unsigned u = __float_as_uint(f);
    return (u & 0x80000000u) ? ~u : (u | 0x80000000u);
}
__device__ __forceinline__ float dec_ordered(unsigned u) {
    return (u & 0x80000000u) ? __uint_as_float(u & 0x7FFFFFFFu)
                             : __uint_as_float(~u);
}

// ------------------------------ tiny utils ---------------------------------

__global__ void zero_u32_kernel(unsigned* __restrict__ p, long long n) {
    long long i = (long long)blockIdx.x * blockDim.x + threadIdx.x;
    if (i < n) p[i] = 0u;
}

__global__ void cvt_bf16_kernel(const float* __restrict__ in,
                                __bf16* __restrict__ out, long long n) {
    long long i = (long long)blockIdx.x * blockDim.x + threadIdx.x;
    if (i < n) out[i] = (__bf16)in[i];
}

// W[K,Nc] (f32, row-major) -> Wt[Nc,K] (bf16)
__global__ void cvt_w_transpose_kernel(const float* __restrict__ W,
                                       __bf16* __restrict__ Wt, int K, int Nc) {
    int i = blockIdx.x * blockDim.x + threadIdx.x;
    if (i >= K * Nc) return;
    int k = i / Nc, n = i % Nc;
    Wt[(size_t)n * K + k] = (__bf16)W[i];
}

// x[i] = f(x[i] + bias[i % C]); f = CELU(alpha=1) if celu!=0 else identity
__global__ void bias_act_kernel(float* __restrict__ x, const float* __restrict__ bias,
                                long long n, int C, int celu) {
    long long i = (long long)blockIdx.x * blockDim.x + threadIdx.x;
    if (i >= n) return;
    float v = x[i] + bias[i % C];
    if (celu) v = v > 0.f ? v : (expf(v) - 1.f);
    x[i] = v;
}

// ------------------------------ WMMA GEMM ----------------------------------
// C[M,Nc] = A_bf16[M,K] @ Bt_bf16[Nc,K]^T + bias[Nc]
// one wave per 16x16 output tile; K multiple of 32, M,Nc multiples of 16.
__global__ void wmma_gemm_bias_kernel(const __bf16* __restrict__ A,
                                      const __bf16* __restrict__ Bt,
                                      const float* __restrict__ bias,
                                      float* __restrict__ C,
                                      int M, int K, int Nc) {
    int wid  = (blockIdx.x * blockDim.x + threadIdx.x) >> 5;
    int lane = threadIdx.x & 31;
    int tilesN = Nc >> 4;
    int tilesM = M >> 4;
    if (wid >= tilesM * tilesN) return;
    int tm = wid / tilesN, tn = wid % tilesN;
    int half = lane >> 4;
    int l15  = lane & 15;
    const size_t arow = (size_t)(tm * 16 + l15) * K;   // A row for this lane
    const size_t brow = (size_t)(tn * 16 + l15) * K;   // Bt row (= B column)

    v8f acc = {};
    for (int kk = 0; kk < K; kk += 32) {
        v16bf a, b;
#pragma unroll
        for (int v = 0; v < 8; ++v) {
            // A 16x32 bf16 layout: V0-3 -> K = 2v + 8*half ; V4-7 -> 16 + 2(v-4) + 8*half
            int ka = kk + ((v < 4) ? (2 * v + 8 * half) : (16 + 2 * (v - 4) + 8 * half));
            a[2 * v]     = A[arow + ka];
            a[2 * v + 1] = A[arow + ka + 1];
            // B 32x16 bf16 layout: same K order as A
            int kb = kk + ((v < 4) ? (2 * v + 8 * half) : (16 + 2 * (v - 4) + 8 * half));
            b[2 * v]     = Bt[brow + kb];
            b[2 * v + 1] = Bt[brow + kb + 1];
        }
        acc = __builtin_amdgcn_wmma_f32_16x16x32_bf16(
                  false, a, false, b, (short)0, acc, false, false);
    }
    int col  = tn * 16 + l15;
    float bv = bias[col];
#pragma unroll
    for (int r = 0; r < 8; ++r) {
        int row = tm * 16 + r + 8 * half;          // C/D layout: row = r + 8*half
        C[(size_t)row * Nc + col] = acc[r] + bv;
    }
}

// ------------------------------ edge kernels -------------------------------
// One wave (32 lanes) per edge; lane = channel (HID == 32).

__global__ void edge_alpha_kernel(const float* __restrict__ xl,
                                  const float* __restrict__ xr,
                                  const float* __restrict__ att,
                                  const long long* __restrict__ ei,
                                  int E, int Etot, int H,
                                  float* __restrict__ alpha,
                                  unsigned* __restrict__ amaxb) {
    int e    = (blockIdx.x * blockDim.x + threadIdx.x) >> 5;
    int lane = threadIdx.x & 31;
    if (e >= Etot) return;
    int s, d; get_edge(ei, E, e, s, d);
    const int HC = H * 32;
    for (int h = 0; h < H; ++h) {
        float vl = xl[(size_t)s * HC + h * 32 + lane];
        float vr = xr[(size_t)d * HC + h * 32 + lane];
        float t  = vl + vr;
        t = t > 0.f ? t : NEG_SLOPE * t;                 // leaky_relu
        float p = t * att[h * 32 + lane];
        for (int off = 16; off > 0; off >>= 1) p += __shfl_xor(p, off, 32);
        if (lane == 0) {
            alpha[(size_t)e * H + h] = p;
            atomicMax(&amaxb[d * H + h], enc_ordered(p));  // segment max
        }
    }
}

__global__ void edge_exp_kernel(float* __restrict__ alpha,       // in-place -> ea
                                const unsigned* __restrict__ amaxb,
                                float* __restrict__ denom,
                                const long long* __restrict__ ei,
                                int E, int Etot, int H) {
    long long i = (long long)blockIdx.x * blockDim.x + threadIdx.x;
    long long tot = (long long)Etot * H;
    if (i >= tot) return;
    int e = (int)(i / H), h = (int)(i % H);
    int s, d; get_edge(ei, E, e, s, d);
    float amax = dec_ordered(amaxb[d * H + h]);
    float ea = expf(alpha[i] - amax);
    alpha[i] = ea;
    unsafeAtomicAdd(&denom[d * H + h], ea);              // segment sum
}

__global__ void edge_aggr_kernel(const float* __restrict__ xl,
                                 const float* __restrict__ ea,
                                 const float* __restrict__ denom,
                                 const long long* __restrict__ ei,
                                 int E, int Etot, int H,
                                 float* __restrict__ out) {
    int e    = (blockIdx.x * blockDim.x + threadIdx.x) >> 5;
    int lane = threadIdx.x & 31;
    if (e >= Etot) return;
    int s, d; get_edge(ei, E, e, s, d);
    const int HC = H * 32;
    for (int h = 0; h < H; ++h) {
        float a = ea[(size_t)e * H + h] / (denom[d * H + h] + 1e-16f);
        float v = xl[(size_t)s * HC + h * 32 + lane] * a;
        unsafeAtomicAdd(&out[(size_t)d * HC + h * 32 + lane], v);
    }
}

// ------------------------------ pooling + head -----------------------------

__global__ void pool_kernel(const float* __restrict__ h2,
                            const long long* __restrict__ batch,
                            float* __restrict__ sums, float* __restrict__ counts,
                            int N) {
    long long i = (long long)blockIdx.x * blockDim.x + threadIdx.x;
    if (i >= (long long)N * 32) return;
    int node = (int)(i >> 5), c = (int)(i & 31);
    int g = (int)batch[node];
    unsafeAtomicAdd(&sums[g * 32 + c], h2[i]);
    if (c == 0) unsafeAtomicAdd(&counts[g], 1.0f);
}

__global__ void head_kernel(const float* __restrict__ sums,
                            const float* __restrict__ counts,
                            const float* __restrict__ Wh1, const float* __restrict__ bh1,
                            const float* __restrict__ Wh2, const float* __restrict__ bh2,
                            float* __restrict__ out, int G) {
    int g = blockIdx.x * blockDim.x + threadIdx.x;
    if (g >= G) return;
    float cnt = fmaxf(counts[g], 1.0f);
    float p[32];
    for (int c = 0; c < 32; ++c) p[c] = sums[g * 32 + c] / cnt;
    float o = bh2[0];
    for (int j = 0; j < 32; ++j) {
        float z = bh1[j];
        for (int c = 0; c < 32; ++c) z += p[c] * Wh1[c * 32 + j];
        z = fmaxf(z, 0.f);
        o += z * Wh2[j];
    }
    out[g] = o;
}

// ------------------------------ launcher -----------------------------------

extern "C" void kernel_launch(void* const* d_in, const int* in_sizes, int n_in,
                              void* d_out, int out_size, void* d_ws, size_t ws_size,
                              hipStream_t stream) {
    const int IN = 64, HID = 32, H1 = 4, G = 256;
    const int N = in_sizes[0] / IN;        // 100000 (divisible by 16)
    const int E = in_sizes[1] / 2;         // 800000
    const int Etot = E + N;                // + self loops

    const float*     x    = (const float*)d_in[0];
    const long long* ei   = (const long long*)d_in[1];
    const long long* batch= (const long long*)d_in[2];
    const float *Wl1=(const float*)d_in[3],  *bl1=(const float*)d_in[4];
    const float *Wr1=(const float*)d_in[5],  *br1=(const float*)d_in[6];
    const float *att1=(const float*)d_in[7], *bias1=(const float*)d_in[8];
    const float *Wl2=(const float*)d_in[9],  *bl2=(const float*)d_in[10];
    const float *Wr2=(const float*)d_in[11], *br2=(const float*)d_in[12];
    const float *att2=(const float*)d_in[13],*bias2=(const float*)d_in[14];
    const float *Wh1=(const float*)d_in[15], *bh1=(const float*)d_in[16];
    const float *Wh2=(const float*)d_in[17], *bh2=(const float*)d_in[18];
    float* out = (float*)d_out;

    // ---- workspace layout (256B aligned regions) ----
    char* ws = (char*)d_ws;
    size_t off = 0;
    auto take = [&](size_t bytes) -> void* {
        void* p = ws + off;
        off += (bytes + 255) & ~(size_t)255;
        return p;
    };
    __bf16*  actbf  = (__bf16*)take((size_t)N * 128 * 2);   // xbf (N x 64) then hbf (N x 128)
    __bf16*  Wl1t   = (__bf16*)take(64 * 128 * 2);
    __bf16*  Wr1t   = (__bf16*)take(64 * 128 * 2);
    __bf16*  Wl2t   = (__bf16*)take(128 * 32 * 2);
    __bf16*  Wr2t   = (__bf16*)take(128 * 32 * 2);
    float*   xl1    = (float*)take((size_t)N * 128 * 4);
    float*   xr1    = (float*)take((size_t)N * 128 * 4);    // reused as out1/h1
    float*   alpha1 = (float*)take((size_t)Etot * H1 * 4);  // reused as ea1
    unsigned*amax1  = (unsigned*)take((size_t)N * H1 * 4);
    float*   denom1 = (float*)take((size_t)N * H1 * 4);
    float*   xl2    = (float*)take((size_t)N * 32 * 4);
    float*   xr2    = (float*)take((size_t)N * 32 * 4);     // reused as out2/h2
    float*   alpha2 = (float*)take((size_t)Etot * 4);
    unsigned*amax2  = (unsigned*)take((size_t)N * 4);
    float*   denom2 = (float*)take((size_t)N * 4);
    float*   sums   = (float*)take(G * 32 * 4);
    float*   counts = (float*)take(G * 4);
    (void)ws_size; (void)n_in; (void)out_size;

    const int TB = 256;                                  // 8 waves / block
    auto blks1d = [](long long n, int tb) { return (int)((n + tb - 1) / tb); };
    const int edgeBlocks = (Etot + 7) / 8;               // wave-per-edge kernels

    // ---- weight conversions (bf16, transposed) ----
    cvt_w_transpose_kernel<<<blks1d(64 * 128, TB), TB, 0, stream>>>(Wl1, Wl1t, IN, H1 * HID);
    cvt_w_transpose_kernel<<<blks1d(64 * 128, TB), TB, 0, stream>>>(Wr1, Wr1t, IN, H1 * HID);
    cvt_w_transpose_kernel<<<blks1d(128 * 32, TB), TB, 0, stream>>>(Wl2, Wl2t, H1 * HID, HID);
    cvt_w_transpose_kernel<<<blks1d(128 * 32, TB), TB, 0, stream>>>(Wr2, Wr2t, H1 * HID, HID);

    // ================= Layer 1 (heads = 4) =================
    cvt_bf16_kernel<<<blks1d((long long)N * IN, TB), TB, 0, stream>>>(x, actbf, (long long)N * IN);
    {   // xl1 = x@Wl1+bl1 ; xr1 = x@Wr1+br1    (M=N, K=64, Nc=128)
        int tiles = (N / 16) * (128 / 16);
        int gb = (tiles + 7) / 8;
        wmma_gemm_bias_kernel<<<gb, TB, 0, stream>>>(actbf, Wl1t, bl1, xl1, N, IN, 128);
        wmma_gemm_bias_kernel<<<gb, TB, 0, stream>>>(actbf, Wr1t, br1, xr1, N, IN, 128);
    }
    zero_u32_kernel<<<blks1d((long long)N * H1, TB), TB, 0, stream>>>(amax1, (long long)N * H1);
    zero_u32_kernel<<<blks1d((long long)N * H1, TB), TB, 0, stream>>>((unsigned*)denom1, (long long)N * H1);
    edge_alpha_kernel<<<edgeBlocks, TB, 0, stream>>>(xl1, xr1, att1, ei, E, Etot, H1, alpha1, amax1);
    // xr1 now free -> becomes aggregation accumulator out1
    zero_u32_kernel<<<blks1d((long long)N * 128, TB), TB, 0, stream>>>((unsigned*)xr1, (long long)N * 128);
    edge_exp_kernel<<<blks1d((long long)Etot * H1, TB), TB, 0, stream>>>(alpha1, amax1, denom1, ei, E, Etot, H1);
    edge_aggr_kernel<<<edgeBlocks, TB, 0, stream>>>(xl1, alpha1, denom1, ei, E, Etot, H1, xr1);
    // h = celu(out1 + bias1)   (in place, xr1 == h1)
    bias_act_kernel<<<blks1d((long long)N * 128, TB), TB, 0, stream>>>(xr1, bias1, (long long)N * 128, 128, 1);

    // ================= Layer 2 (heads = 1) =================
    cvt_bf16_kernel<<<blks1d((long long)N * 128, TB), TB, 0, stream>>>(xr1, actbf, (long long)N * 128);
    {   // xl2/xr2 = h@W + b   (M=N, K=128, Nc=32)
        int tiles = (N / 16) * (32 / 16);
        int gb = (tiles + 7) / 8;
        wmma_gemm_bias_kernel<<<gb, TB, 0, stream>>>(actbf, Wl2t, bl2, xl2, N, 128, HID);
        wmma_gemm_bias_kernel<<<gb, TB, 0, stream>>>(actbf, Wr2t, br2, xr2, N, 128, HID);
    }
    zero_u32_kernel<<<blks1d(N, TB), TB, 0, stream>>>(amax2, N);
    zero_u32_kernel<<<blks1d(N, TB), TB, 0, stream>>>((unsigned*)denom2, N);
    edge_alpha_kernel<<<edgeBlocks, TB, 0, stream>>>(xl2, xr2, att2, ei, E, Etot, 1, alpha2, amax2);
    zero_u32_kernel<<<blks1d((long long)N * 32, TB), TB, 0, stream>>>((unsigned*)xr2, (long long)N * 32);
    edge_exp_kernel<<<blks1d((long long)Etot, TB), TB, 0, stream>>>(alpha2, amax2, denom2, ei, E, Etot, 1);
    edge_aggr_kernel<<<edgeBlocks, TB, 0, stream>>>(xl2, alpha2, denom2, ei, E, Etot, 1, xr2);
    bias_act_kernel<<<blks1d((long long)N * 32, TB), TB, 0, stream>>>(xr2, bias2, (long long)N * 32, 32, 0);

    // ================= mean pool + head =================
    zero_u32_kernel<<<blks1d(G * 32 + G, TB), TB, 0, stream>>>((unsigned*)sums, G * 32 + G); // sums+counts contiguous
    pool_kernel<<<blks1d((long long)N * 32, TB), TB, 0, stream>>>(xr2, batch, sums, counts, N);
    head_kernel<<<1, G, 0, stream>>>(sums, counts, Wh1, bh1, Wh2, bh2, out, G);
}
